// Block_73796128080521
// MI455X (gfx1250) — matrix operand order repaired and checked by reading, not compiled, not run on hardware
//
#include <hip/hip_runtime.h>

typedef _Float16 v16h __attribute__((ext_vector_type(16)));
typedef _Float16 h8   __attribute__((ext_vector_type(8)));
typedef float    v8f  __attribute__((ext_vector_type(8)));

// ---------------- WMMA fragment helpers (contiguous k = 16*(lane/16) + s, same in A and B) -----

// A-matrix 16x32 f16, row-major source with leading dim ld (in elements).
// lane L: m = L&15, khalf = L>>4.  VGPR0-7 hold 16 contiguous K
// starting at khalf*16.
__device__ inline v16h load_frag_a(const _Float16* p, int ld) {
    int lane = threadIdx.x & 31;
    int m = lane & 15, kh = (lane >> 4) & 1;
    const _Float16* q = p + (size_t)m * ld + kh * 16;
    h8 lo = *(const h8*)(q);
    h8 hi = *(const h8*)(q + 8);
    v16h f;
#pragma unroll
    for (int i = 0; i < 8; i++) { f[i] = lo[i]; f[i + 8] = hi[i]; }
    return f;
}

// B-matrix 32x16 f16 (KxN). Stored as [N][K] row-major (torch Linear weight
// layout) with leading dim ld. lane L: n = L&15, holds 16 contiguous K
// starting at (L>>4)*16.
__device__ inline v16h load_frag_b(const _Float16* p, int ld) {
    int lane = threadIdx.x & 31;
    int n = lane & 15, kh = (lane >> 4) & 1;
    return *(const v16h*)(p + (size_t)n * ld + kh * 16);
}

__device__ inline v8f wmma_f16(v16h a, v16h b, v8f c) {
    return __builtin_amdgcn_wmma_f32_16x16x32_f16(false, a, false, b,
                                                  (short)0, c, false, false);
}

// C/D element (m,n): vgpr r -> m = (lane>>4)*8 + r, n = lane&15.

// ---------------- constants -------------------------------------------------
#define Bb   8
#define Hh   56
#define Cc   384
#define Ll   (Hh * Hh)        // 3136
#define TOK  (Bb * Ll)        // 25088
#define NH   12
#define HD   32
#define WS7  7
#define NN   49
#define NWIN 512              // B * 8 * 8
#define QKLD 768
#define C4   1536

// workspace offsets (bytes), all 256B-aligned
#define OFF_WQKV  ((size_t)0)                       // 1152*384 f16
#define OFF_WPROJ (OFF_WQKV + 884736)               // 384*384 f16
#define OFF_WFC1  (OFF_WPROJ + 294912)              // 1536*384 f16
#define OFF_WFC2  (OFF_WFC1 + 1179648)              // 384*1536 f16
#define OFF_POOL  (OFF_WFC2 + 1179648)              // 3538944
#define OFF_XW    (OFF_POOL)                        // 25088*384 f16
#define OFF_QK    (OFF_XW + 19267584)               // (25088+16)*768 f16
#define OFF_VT    (OFF_QK + 38559744)               // 512*12*32*64 f16 (zeroed)
#define POOL_END  (OFF_VT + 25165824)
#define OFF_FC1O  (OFF_POOL)                        // alias: 25088*1536 f16
#define OFF_AO    (POOL_END)                        // 25088*384 f16
#define OFF_H2    (OFF_AO)                          // alias (ao dead after proj)
#define OFF_X1    (OFF_AO + 19267584)               // 25088*384 f32

// ---------------- kernels ---------------------------------------------------

__global__ __launch_bounds__(256) void cvt_f16(const float* __restrict__ s,
                                               _Float16* __restrict__ d, int n) {
    int i = blockIdx.x * 256 + threadIdx.x;
    if (i < n) d[i] = (_Float16)s[i];
}

// LayerNorm over C=384; one wave per token. windowed=1 -> write in
// window-major row order for the attention path.
__global__ __launch_bounds__(256) void ln_kernel(const float* __restrict__ x,
                                                 const float* __restrict__ g,
                                                 const float* __restrict__ b,
                                                 _Float16* __restrict__ out,
                                                 int windowed) {
    int lane = threadIdx.x & 31;
    int wave = threadIdx.x >> 5;
    int t = blockIdx.x * 8 + wave;              // token id, 0..25087
    const float* row = x + (size_t)t * Cc;
    float vals[12];
    float s = 0.f, ss = 0.f;
#pragma unroll
    for (int i = 0; i < 12; i++) {
        float v = row[i * 32 + lane];
        vals[i] = v; s += v; ss += v * v;
    }
#pragma unroll
    for (int o = 16; o > 0; o >>= 1) {
        s  += __shfl_xor(s,  o, 32);
        ss += __shfl_xor(ss, o, 32);
    }
    float mean = s * (1.f / Cc);
    float var  = ss * (1.f / Cc) - mean * mean;
    float inv  = rsqrtf(var + 1e-5f);
    size_t orow;
    if (windowed) {
        int bb = t / Ll, l = t % Ll;
        int y = l / Hh, xx = l % Hh;
        int wy = y / WS7, iy = y % WS7, wx = xx / WS7, ix = xx % WS7;
        orow = (size_t)((bb * 64 + wy * 8 + wx) * NN + iy * WS7 + ix);
    } else {
        orow = (size_t)t;
    }
#pragma unroll
    for (int i = 0; i < 12; i++) {
        int c = i * 32 + lane;
        out[orow * Cc + c] = (_Float16)((vals[i] - mean) * inv * g[c] + b[c]);
    }
}

// 32x64-per-wave GEMM core; block tile 64x128 with 4 waves (2x2).
// Per K-step: 2 A-frags + 4 B-frags (12 b128 loads) feed 8 WMMAs.
__device__ inline void gemm_core(const _Float16* __restrict__ A, int lda,
                                 const _Float16* __restrict__ B, int ldb,
                                 int K, int m0, int n0, v8f acc[2][4]) {
    for (int k = 0; k < K; k += 32) {
        v16h a0 = load_frag_a(A + (size_t)m0 * lda + k, lda);
        v16h a1 = load_frag_a(A + (size_t)(m0 + 16) * lda + k, lda);
        v16h b0 = load_frag_b(B + (size_t)n0 * ldb + k, ldb);
        v16h b1 = load_frag_b(B + (size_t)(n0 + 16) * ldb + k, ldb);
        v16h b2 = load_frag_b(B + (size_t)(n0 + 32) * ldb + k, ldb);
        v16h b3 = load_frag_b(B + (size_t)(n0 + 48) * ldb + k, ldb);
        acc[0][0] = wmma_f16(a0, b0, acc[0][0]);
        acc[0][1] = wmma_f16(a0, b1, acc[0][1]);
        acc[0][2] = wmma_f16(a0, b2, acc[0][2]);
        acc[0][3] = wmma_f16(a0, b3, acc[0][3]);
        acc[1][0] = wmma_f16(a1, b0, acc[1][0]);
        acc[1][1] = wmma_f16(a1, b1, acc[1][1]);
        acc[1][2] = wmma_f16(a1, b2, acc[1][2]);
        acc[1][3] = wmma_f16(a1, b3, acc[1][3]);
    }
}

// wave/block coordinate helper: 128 threads = 4 waves arranged 2(M) x 2(N)
#define GEMM_COORDS                                            \
    int lane = threadIdx.x & 31;                               \
    int wave = threadIdx.x >> 5;                               \
    int m0 = blockIdx.x * 64 + (wave >> 1) * 32;               \
    int n0 = blockIdx.y * 128 + (wave & 1) * 64;               \
    int nb = lane & 15, mB = (lane >> 4) * 8;

// QKV: [25088,384] x [1152,384]^T. cols<768 -> qk row-major; cols>=768 -> V^T.
__global__ __launch_bounds__(128) void gemm_qkv(const _Float16* __restrict__ A,
                                                const _Float16* __restrict__ B,
                                                _Float16* __restrict__ qk,
                                                _Float16* __restrict__ vT) {
    GEMM_COORDS
    v8f acc[2][4] = {};
    gemm_core(A, Cc, B, Cc, Cc, m0, n0, acc);
#pragma unroll
    for (int ti = 0; ti < 2; ti++)
#pragma unroll
    for (int tj = 0; tj < 4; tj++)
#pragma unroll
    for (int r = 0; r < 8; r++) {
        int m = m0 + ti * 16 + mB + r;
        int n = n0 + tj * 16 + nb;
        float v = acc[ti][tj][r];
        if (n < 768) {
            qk[(size_t)m * QKLD + n] = (_Float16)v;
        } else {
            int h = (n - 768) >> 5, d = (n - 768) & 31;
            int bw = m / NN, mm = m % NN;
            vT[(size_t)((bw * NH + h) * HD + d) * 64 + mm] = (_Float16)v;
        }
    }
}

// attention per (window, head): S=QK^T (WMMA), +bias, mask to 64, softmax, P·V
__global__ __launch_bounds__(128) void attn_kernel(
        const _Float16* __restrict__ qk, const _Float16* __restrict__ vT,
        const float* __restrict__ bias_tab, _Float16* __restrict__ ao) {
    __shared__ __align__(16) float    S[64 * 64];
    __shared__ __align__(16) _Float16 P[64 * 64];
    int bw = blockIdx.x / NH;
    int h  = blockIdx.x % NH;
    int wave = threadIdx.x >> 5, lane = threadIdx.x & 31;
    int nb = lane & 15, mB = (lane >> 4) * 8;
    int m0 = wave * 16;

    const _Float16* qbase = qk + (size_t)(bw * NN) * QKLD + h * HD;
    const _Float16* kbase = qbase + Cc;       // k cols start at 384
    v16h a = load_frag_a(qbase + (size_t)m0 * QKLD, QKLD);
    const float scale = 0.17677669529663687f; // 1/sqrt(32)
#pragma unroll
    for (int j = 0; j < 4; j++) {
        v16h bfrag = load_frag_b(kbase + (size_t)(j * 16) * QKLD, QKLD);
        v8f acc = {};
        acc = wmma_f16(a, bfrag, acc);
#pragma unroll
        for (int r = 0; r < 8; r++) {
            int m = m0 + mB + r;
            int n = j * 16 + nb;
            float val = -1e30f;
            if (m < NN && n < NN) {
                int iy = m / WS7, ix = m % WS7, jy = n / WS7, jx = n % WS7;
                int idx = (iy - jy + 6) * 13 + (ix - jx + 6);
                val = acc[r] * scale + bias_tab[idx * NH + h];
            }
            S[m * 64 + n] = val;
        }
    }
    __syncthreads();
    int row = threadIdx.x;
    if (row < 64) {
        float mx = -3.4e38f;
        for (int n = 0; n < 64; n++) mx = fmaxf(mx, S[row * 64 + n]);
        float sum = 0.f;
        for (int n = 0; n < 64; n++) {
            float e = __expf(S[row * 64 + n] - mx);
            S[row * 64 + n] = e; sum += e;
        }
        float inv = 1.f / sum;
        for (int n = 0; n < 64; n++)
            P[row * 64 + n] = (_Float16)(S[row * 64 + n] * inv);
    }
    __syncthreads();
    const _Float16* vb = vT + (size_t)((bw * NH + h) * HD) * 64;
    v8f o0 = {}, o1 = {};
#pragma unroll
    for (int k0 = 0; k0 < 64; k0 += 32) {
        v16h pa = load_frag_a(&P[m0 * 64 + k0], 64);
        v16h b0 = load_frag_b(vb + k0, 64);
        v16h b1 = load_frag_b(vb + (size_t)16 * 64 + k0, 64);
        o0 = wmma_f16(pa, b0, o0);
        o1 = wmma_f16(pa, b1, o1);
    }
#pragma unroll
    for (int r = 0; r < 8; r++) {
        int m = m0 + mB + r;
        if (m < NN) {
            size_t orow = (size_t)(bw * NN + m) * Cc + h * HD;
            ao[orow + nb]      = (_Float16)o0[r];
            ao[orow + 16 + nb] = (_Float16)o1[r];
        }
    }
}

// proj + window-reverse + x1 = x + ls1 * (o W^T + b)
__global__ __launch_bounds__(128) void gemm_proj(const _Float16* __restrict__ A,
                                                 const _Float16* __restrict__ B,
                                                 const float* __restrict__ bias,
                                                 const float* __restrict__ xin,
                                                 const float* __restrict__ ls1,
                                                 float* __restrict__ x1) {
    GEMM_COORDS
    v8f acc[2][4] = {};
    gemm_core(A, Cc, B, Cc, Cc, m0, n0, acc);
#pragma unroll
    for (int ti = 0; ti < 2; ti++)
#pragma unroll
    for (int tj = 0; tj < 4; tj++)
#pragma unroll
    for (int r = 0; r < 8; r++) {
        int m = m0 + ti * 16 + mB + r;
        int n = n0 + tj * 16 + nb;
        int bw = m / NN, mm = m % NN;
        int bimg = bw >> 6, wy = (bw >> 3) & 7, wx = bw & 7;
        int iy = mm / WS7, ix = mm % WS7;
        int l = (wy * WS7 + iy) * Hh + wx * WS7 + ix;
        size_t t = (size_t)bimg * Ll + l;
        float v = acc[ti][tj][r] + bias[n];
        x1[t * Cc + n] = xin[t * Cc + n] + ls1[n] * v;
    }
}

// fc1 + exact GELU
__global__ __launch_bounds__(128) void gemm_fc1(const _Float16* __restrict__ A,
                                                const _Float16* __restrict__ B,
                                                const float* __restrict__ bias,
                                                _Float16* __restrict__ out) {
    GEMM_COORDS
    v8f acc[2][4] = {};
    gemm_core(A, Cc, B, Cc, Cc, m0, n0, acc);
#pragma unroll
    for (int ti = 0; ti < 2; ti++)
#pragma unroll
    for (int tj = 0; tj < 4; tj++)
#pragma unroll
    for (int r = 0; r < 8; r++) {
        int m = m0 + ti * 16 + mB + r;
        int n = n0 + tj * 16 + nb;
        float t = acc[ti][tj][r] + bias[n];
        float g = 0.5f * t * (1.f + erff(t * 0.70710678118654752f));
        out[(size_t)m * C4 + n] = (_Float16)g;
    }
}

// fc2 + final residual: out = x1 + ls2 * (h W^T + b)
__global__ __launch_bounds__(128) void gemm_fc2(const _Float16* __restrict__ A,
                                                const _Float16* __restrict__ B,
                                                const float* __restrict__ bias,
                                                const float* __restrict__ x1,
                                                const float* __restrict__ ls2,
                                                float* __restrict__ out) {
    GEMM_COORDS
    v8f acc[2][4] = {};
    gemm_core(A, C4, B, C4, C4, m0, n0, acc);
#pragma unroll
    for (int ti = 0; ti < 2; ti++)
#pragma unroll
    for (int tj = 0; tj < 4; tj++)
#pragma unroll
    for (int r = 0; r < 8; r++) {
        int m = m0 + ti * 16 + mB + r;
        int n = n0 + tj * 16 + nb;
        float v = acc[ti][tj][r] + bias[n];
        out[(size_t)m * Cc + n] = x1[(size_t)m * Cc + n] + ls2[n] * v;
    }
}

// ---------------- launch ----------------------------------------------------
extern "C" void kernel_launch(void* const* d_in, const int* in_sizes, int n_in,
                              void* d_out, int out_size, void* d_ws, size_t ws_size,
                              hipStream_t stream) {
    const float* x       = (const float*)d_in[0];
    const float* q_w     = (const float*)d_in[1];
    const float* kv_w    = (const float*)d_in[2];
    const float* proj_w  = (const float*)d_in[3];
    const float* proj_b  = (const float*)d_in[4];
    const float* fc1_w   = (const float*)d_in[5];
    const float* fc1_b   = (const float*)d_in[6];
    const float* fc2_w   = (const float*)d_in[7];
    const float* fc2_b   = (const float*)d_in[8];
    const float* norm1_g = (const float*)d_in[9];
    const float* norm1_b = (const float*)d_in[10];
    const float* norm2_g = (const float*)d_in[11];
    const float* norm2_b = (const float*)d_in[12];
    const float* ls1     = (const float*)d_in[13];
    const float* ls2     = (const float*)d_in[14];
    const float* relb    = (const float*)d_in[15];
    float* outp = (float*)d_out;

    char* ws = (char*)d_ws;
    _Float16* wqkv = (_Float16*)(ws + OFF_WQKV);
    _Float16* wprj = (_Float16*)(ws + OFF_WPROJ);
    _Float16* wf1  = (_Float16*)(ws + OFF_WFC1);
    _Float16* wf2  = (_Float16*)(ws + OFF_WFC2);
    _Float16* xw   = (_Float16*)(ws + OFF_XW);
    _Float16* qkb  = (_Float16*)(ws + OFF_QK);
    _Float16* vtb  = (_Float16*)(ws + OFF_VT);
    _Float16* ao   = (_Float16*)(ws + OFF_AO);
    _Float16* h2   = (_Float16*)(ws + OFF_H2);
    _Float16* f1o  = (_Float16*)(ws + OFF_FC1O);
    float*    x1   = (float*)(ws + OFF_X1);

    // weights -> f16 (q and kv concatenated into one [1152,384] matrix)
    cvt_f16<<<(147456 + 255) / 256, 256, 0, stream>>>(q_w,  wqkv, 147456);
    cvt_f16<<<(294912 + 255) / 256, 256, 0, stream>>>(kv_w, wqkv + 147456, 294912);
    cvt_f16<<<(147456 + 255) / 256, 256, 0, stream>>>(proj_w, wprj, 147456);
    cvt_f16<<<(589824 + 255) / 256, 256, 0, stream>>>(fc1_w, wf1, 589824);
    cvt_f16<<<(589824 + 255) / 256, 256, 0, stream>>>(fc2_w, wf2, 589824);

    // zero V^T (pad tokens 49..63 must be exactly 0)
    hipMemsetAsync(ws + OFF_VT, 0, (size_t)25165824, stream);

    // LN1 + window partition
    ln_kernel<<<TOK / 8, 256, 0, stream>>>(x, norm1_g, norm1_b, xw, 1);

    // QKV projection
    gemm_qkv<<<dim3(TOK / 64, 9), 128, 0, stream>>>(xw, wqkv, qkb, vtb);

    // windowed attention
    attn_kernel<<<NWIN * NH, 128, 0, stream>>>(qkb, vtb, relb, ao);

    // proj + window reverse + layer-scale residual
    gemm_proj<<<dim3(TOK / 64, 3), 128, 0, stream>>>(ao, wprj, proj_b, x, ls1, x1);

    // LN2
    ln_kernel<<<TOK / 8, 256, 0, stream>>>(x1, norm2_g, norm2_b, h2, 0);

    // MLP
    gemm_fc1<<<dim3(TOK / 64, 12), 128, 0, stream>>>(h2, wf1, fc1_b, f1o);
    gemm_fc2<<<dim3(TOK / 64, 3), 128, 0, stream>>>(f1o, wf2, fc2_b, x1, ls2, outp);
}
